// ProfiledLlamaAttention_22883585753578
// MI455X (gfx1250) — matrix operand order repaired and checked by reading, not compiled, not run on hardware
//
#include <hip/hip_runtime.h>
#include <hip/hip_bf16.h>

typedef unsigned short u16;
typedef unsigned int   u32;

typedef __bf16 v16bf __attribute__((ext_vector_type(16)));
typedef float  v8f   __attribute__((ext_vector_type(8)));

#define N_HEADS   16
#define HEAD_DIM  128
#define BATCH     2
#define SEQ       2048
#define DMODEL    2048
#define MROWS     (BATCH * SEQ)                // 4096
#define NEG_INF   (-1000000000.0f)
#define INV_SQRT_HD 0.08838834764831845f       // 1/sqrt(128)
#define CAP_INV   0.02f                        // 1/50
#define CAP       50.0f

// ---------- bf16 helpers ----------
__device__ __forceinline__ u16 f2bf(float f) {
    u32 u = __float_as_uint(f);
    u32 r = u + 0x7fffu + ((u >> 16) & 1u);     // round-to-nearest-even
    return (u16)(r >> 16);
}
__device__ __forceinline__ float bf2f(u16 h) {
    return __uint_as_float(((u32)h) << 16);
}

// Load a 32-byte WMMA operand fragment as two 16-byte chunks and bit-cast.
struct Frag32 { uint4 lo, hi; };
__device__ __forceinline__ v16bf ld_frag(const u16* plo, const u16* phi) {
    Frag32 t;
    t.lo = *(const uint4*)plo;
    t.hi = *(const uint4*)phi;
    return __builtin_bit_cast(v16bf, t);
}

// ---------- fp32 -> bf16 conversion ----------
__global__ void cvt_bf16_kernel(const float* __restrict__ src, u16* __restrict__ dst, int n) {
    int i = blockIdx.x * blockDim.x + threadIdx.x;
    int stride = gridDim.x * blockDim.x;
    for (; i < n; i += stride) dst[i] = f2bf(src[i]);
}

// ---------- Tiled bf16 WMMA GEMM:  C[M,N] = A[M,K] * W[N,K]^T ----------
// BLOCK 128x128, BK=32, 256 threads = 8 waves in 4(M) x 2(N), wave tile 32x64.
// mode 0: store bf16 transposed to [B,H,S,HD] (QKV path, dst picked by blockIdx.z)
// mode 1: store fp32 row-major to outf
#define LDS_STR 40   // 32 + 8 pad bf16 elems -> 80B row stride: conflict-free b128 frag reads

__global__ __launch_bounds__(256)
void gemm_bf16_kernel(const u16* __restrict__ A,
                      const u16* __restrict__ w0, const u16* __restrict__ w1, const u16* __restrict__ w2,
                      u16* __restrict__ d0, u16* __restrict__ d1, u16* __restrict__ d2,
                      float* __restrict__ outf, int mode)
{
    __shared__ u16 As[128 * LDS_STR];
    __shared__ u16 Bs[128 * LDS_STR];

    const int tid  = threadIdx.x;
    const int lane = tid & 31;
    const int wv   = tid >> 5;
    const int wm   = wv >> 1;            // 0..3 -> m offset wm*32
    const int wn   = wv & 1;             // 0..1 -> n offset wn*64
    const int hr   = lane & 15;
    const int half = lane >> 4;

    const int z = blockIdx.z;
    const u16* W   = (z == 0) ? w0 : (z == 1) ? w1 : w2;
    u16*       Dst = (z == 0) ? d0 : (z == 1) ? d1 : d2;

    const int mBase = blockIdx.y * 128;
    const int nBase = blockIdx.x * 128;

    v8f acc[2][4];
#pragma unroll
    for (int mi = 0; mi < 2; ++mi)
#pragma unroll
        for (int ni = 0; ni < 4; ++ni) acc[mi][ni] = (v8f)0.0f;

    const int r = tid >> 2;              // 0..63
    const int c = (tid & 3) * 8;         // 0,8,16,24

    for (int k0 = 0; k0 < DMODEL; k0 += 32) {
        // cooperative load of A-tile and W-tile into LDS (16B per load, coalesced)
#pragma unroll
        for (int p = 0; p < 2; ++p) {
            int row = r + p * 64;
            uint4 va = *(const uint4*)&A[(size_t)(mBase + row) * DMODEL + k0 + c];
            *(uint4*)&As[row * LDS_STR + c] = va;
            uint4 vb = *(const uint4*)&W[(size_t)(nBase + row) * DMODEL + k0 + c];
            *(uint4*)&Bs[row * LDS_STR + c] = vb;
        }
        if (k0 + 32 < DMODEL) {          // prefetch next tiles (global_prefetch_b8)
            __builtin_prefetch(&A[(size_t)(mBase + r) * DMODEL + k0 + 32 + c], 0, 0);
            __builtin_prefetch(&W[(size_t)(nBase + r) * DMODEL + k0 + 32 + c], 0, 0);
        }
        __syncthreads();

        // A fragment: lane holds row (hr), K = half*8..+7 then 16+half*8..+7
        v16bf af[2], bf[4];
#pragma unroll
        for (int mi = 0; mi < 2; ++mi) {
            const u16* base = &As[(wm * 32 + mi * 16 + hr) * LDS_STR];
            af[mi] = ld_frag(base + half * 8, base + 16 + half * 8);
        }
        // B fragment: lane holds col (hr), K = half*8..+7 then 16+half*8..+7
#pragma unroll
        for (int ni = 0; ni < 4; ++ni) {
            const u16* base = &Bs[(wn * 64 + ni * 16 + hr) * LDS_STR];
            bf[ni] = ld_frag(base + half * 8, base + 16 + half * 8);
        }
#pragma unroll
        for (int mi = 0; mi < 2; ++mi)
#pragma unroll
            for (int ni = 0; ni < 4; ++ni)
                acc[mi][ni] = __builtin_amdgcn_wmma_f32_16x16x32_bf16(
                    false, af[mi], false, bf[ni], (short)0, acc[mi][ni], false, false);
        __syncthreads();
    }

    // store: D layout = lane holds col hr, VGPR i holds row half*8+i
#pragma unroll
    for (int mi = 0; mi < 2; ++mi)
#pragma unroll
        for (int ni = 0; ni < 4; ++ni)
#pragma unroll
            for (int i = 0; i < 8; ++i) {
                int m = mBase + wm * 32 + mi * 16 + half * 8 + i;
                int n = nBase + wn * 64 + ni * 16 + hr;
                if (mode == 0) {
                    int b_ = m >> 11, s = m & 2047;
                    int h_ = n >> 7,  d = n & 127;
                    Dst[(size_t)((b_ * N_HEADS + h_) * SEQ + s) * HEAD_DIM + d] = f2bf(acc[mi][ni][i]);
                } else {
                    outf[(size_t)m * DMODEL + n] = acc[mi][ni][i];
                }
            }
}

// ---------- RoPE in-place on bf16 Q/K laid out [B,H,S,HD] ----------
__global__ void rope_kernel(u16* __restrict__ Qb, u16* __restrict__ Kb,
                            const float* __restrict__ fc)
{
    const int NP = BATCH * N_HEADS * SEQ * (HEAD_DIM / 2);   // pairs per array
    int i = blockIdx.x * blockDim.x + threadIdx.x;
    int stride = gridDim.x * blockDim.x;
    for (; i < 2 * NP; i += stride) {
        u16* arr = (i < NP) ? Qb : Kb;
        int idx  = (i < NP) ? i : i - NP;
        int ip = idx & 63;                 // d/2
        int s  = (idx >> 6) & (SEQ - 1);
        int bh = idx >> 17;                // 0..31
        u32* p = (u32*)arr + ((size_t)(bh * SEQ + s) * (HEAD_DIM / 2) + ip);
        u32 w = *p;
        float a = bf2f((u16)(w & 0xffff));
        float b = bf2f((u16)(w >> 16));
        float co = fc[s * 128 + ip * 2];
        float si = fc[s * 128 + ip * 2 + 1];
        float o0 = a * co - b * si;
        float o1 = a * si + b * co;
        *p = (u32)f2bf(o0) | ((u32)f2bf(o1) << 16);
    }
}

// ---------- Flash attention with Gemma2 tanh softcap, causal ----------
// grid: (S/128, B*H). 8 waves; wave w owns 16 q-rows. Key blocks of 64.
#define KS_STR 136     // 128 + 8 pad
#define VT_STR 72      // 64 + 8 pad
#define PW_STR 72      // 64 + 8 pad

__global__ __launch_bounds__(256)
void attn_kernel(const u16* __restrict__ Q, const u16* __restrict__ K,
                 const u16* __restrict__ V, u16* __restrict__ O)
{
    __shared__ u16 Ks[64 * KS_STR];          // [key][d]
    __shared__ u16 Vt[128 * VT_STR];         // [d][key]  (transposed)
    __shared__ u16 Pw[8 * 16 * PW_STR];      // wave-private P staging

    const int tid  = threadIdx.x;
    const int lane = tid & 31;
    const int wv   = tid >> 5;
    const int hr   = lane & 15;
    const int half = lane >> 4;

    const int bh = blockIdx.y;
    const int qb = blockIdx.x;
    const int q0 = qb * 128 + wv * 16;       // this wave's first q row

    // Q fragments resident in registers: 16 rows x 128 d = 4 A-frags
    const u16* Qrow = Q + (size_t)(bh * SEQ + q0 + hr) * HEAD_DIM;
    v16bf qa[4];
#pragma unroll
    for (int kk = 0; kk < 4; ++kk)
        qa[kk] = ld_frag(Qrow + kk * 32 + half * 8, Qrow + kk * 32 + 16 + half * 8);

    v8f o[8];
#pragma unroll
    for (int nt = 0; nt < 8; ++nt) o[nt] = (v8f)0.0f;
    float mrow[8], lrow[8];
#pragma unroll
    for (int i = 0; i < 8; ++i) { mrow[i] = -1e30f; lrow[i] = 0.0f; }

    const int nkb = qb * 2 + 2;              // causal: key blocks needed by this q-block
    for (int kb = 0; kb < nkb; ++kb) {
        const int k0 = kb * 64;
        __syncthreads();                      // protect LDS reuse

        // K block -> LDS, row-major (1024 x 16B chunks)
        const u16* Kg = K + (size_t)(bh * SEQ + k0) * HEAD_DIM;
#pragma unroll
        for (int it = 0; it < 4; ++it) {
            int ch = tid + it * 256;
            int r  = ch >> 3;
            int c  = (ch & 7) * 8;
            *(uint4*)&Ks[r * KS_STR + c] = *(const uint4*)&Kg[r * HEAD_DIM + c];
        }
        // V block -> LDS transposed: Vt[d][key]
        const u32* Vg = (const u32*)(V + (size_t)(bh * SEQ + k0) * HEAD_DIM);
#pragma unroll
        for (int it = 0; it < 16; ++it) {
            int j   = tid + it * 256;        // dword index over 64x64 dwords
            int key = j >> 6;
            int dp  = j & 63;
            u32 w = Vg[key * 64 + dp];
            Vt[(dp * 2) * VT_STR + key]     = (u16)(w & 0xffff);
            Vt[(dp * 2 + 1) * VT_STR + key] = (u16)(w >> 16);
        }
        __syncthreads();

        if (k0 <= q0 + 15) {                 // skip blocks entirely above this wave's rows
            // S = Q * K^T  (16 q-rows x 64 keys)
            v8f s[4];
#pragma unroll
            for (int ni = 0; ni < 4; ++ni) {
                s[ni] = (v8f)0.0f;
#pragma unroll
                for (int kk = 0; kk < 4; ++kk) {
                    const u16* pb = &Ks[(ni * 16 + hr) * KS_STR + kk * 32];
                    s[ni] = __builtin_amdgcn_wmma_f32_16x16x32_bf16(
                        false, qa[kk], false, ld_frag(pb + half * 8, pb + 16 + half * 8), (short)0, s[ni], false, false);
                }
            }
            // scale + tanh softcap + causal mask
            float vals[4][8];
            float rmax[8];
#pragma unroll
            for (int i = 0; i < 8; ++i) rmax[i] = -1e30f;
#pragma unroll
            for (int ni = 0; ni < 4; ++ni)
#pragma unroll
                for (int i = 0; i < 8; ++i) {
                    float x = s[ni][i] * INV_SQRT_HD;
                    float y = fminf(fmaxf(x * CAP_INV, -15.0f), 15.0f);
                    float e2 = __expf(2.0f * y);
                    float t = ((e2 - 1.0f) / (e2 + 1.0f)) * CAP;
                    int key = k0 + ni * 16 + hr;
                    int qq  = q0 + half * 8 + i;
                    t += (key <= qq) ? 0.0f : NEG_INF;
                    vals[ni][i] = t;
                    rmax[i] = fmaxf(rmax[i], t);
                }
            // row-max across the 16 lanes of each half
#pragma unroll
            for (int i = 0; i < 8; ++i)
#pragma unroll
                for (int msk = 1; msk < 16; msk <<= 1)
                    rmax[i] = fmaxf(rmax[i], __shfl_xor(rmax[i], msk, 32));

            float rsum[8];
#pragma unroll
            for (int i = 0; i < 8; ++i) {
                float mnew  = fmaxf(mrow[i], rmax[i]);
                float alpha = __expf(mrow[i] - mnew);
                mrow[i] = mnew;
                float sum = 0.0f;
#pragma unroll
                for (int ni = 0; ni < 4; ++ni) {
                    float p = __expf(vals[ni][i] - mnew);
                    vals[ni][i] = p;
                    sum += p;
                }
                rsum[i] = sum;
                lrow[i] *= alpha;
#pragma unroll
                for (int nt = 0; nt < 8; ++nt) o[nt][i] *= alpha;
            }
#pragma unroll
            for (int i = 0; i < 8; ++i) {
#pragma unroll
                for (int msk = 1; msk < 16; msk <<= 1)
                    rsum[i] += __shfl_xor(rsum[i], msk, 32);
                lrow[i] += rsum[i];
            }
            // stage P (C-layout) to wave-private LDS, reload in A-layout (same-wave LDS in-order)
            u16* Pm = &Pw[wv * 16 * PW_STR];
#pragma unroll
            for (int ni = 0; ni < 4; ++ni)
#pragma unroll
                for (int i = 0; i < 8; ++i)
                    Pm[(half * 8 + i) * PW_STR + ni * 16 + hr] = f2bf(vals[ni][i]);

            v16bf pa[2];
#pragma unroll
            for (int kk2 = 0; kk2 < 2; ++kk2) {
                const u16* base = &Pm[hr * PW_STR] + kk2 * 32;
                pa[kk2] = ld_frag(base + half * 8, base + 16 + half * 8);
            }
            // O += P * V  (16 x 64 x 128)
#pragma unroll
            for (int nt = 0; nt < 8; ++nt)
#pragma unroll
                for (int kk2 = 0; kk2 < 2; ++kk2) {
                    const u16* pv = &Vt[(nt * 16 + hr) * VT_STR + kk2 * 32];
                    o[nt] = __builtin_amdgcn_wmma_f32_16x16x32_bf16(
                        false, pa[kk2], false, ld_frag(pv + half * 8, pv + 16 + half * 8), (short)0, o[nt], false, false);
                }
        }
    }

    // normalize + store bf16 to [B,S,H*HD]
    const int b_ = bh >> 4, h_ = bh & 15;
#pragma unroll
    for (int i = 0; i < 8; ++i) {
        float inv = 1.0f / lrow[i];
        int qq = q0 + half * 8 + i;
#pragma unroll
        for (int nt = 0; nt < 8; ++nt) {
            int d = nt * 16 + hr;
            O[(size_t)(b_ * SEQ + qq) * DMODEL + h_ * HEAD_DIM + d] = f2bf(o[nt][i] * inv);
        }
    }
}

// ---------- host launch ----------
extern "C" void kernel_launch(void* const* d_in, const int* in_sizes, int n_in,
                              void* d_out, int out_size, void* d_ws, size_t ws_size,
                              hipStream_t stream)
{
    (void)in_sizes; (void)n_in; (void)out_size; (void)ws_size;
    const float* x   = (const float*)d_in[0];
    const float* fc  = (const float*)d_in[1];
    // d_in[2] is the causal mask; we synthesize it in-kernel (identical -1e9 fill).
    const float* wq  = (const float*)d_in[3];
    const float* wk  = (const float*)d_in[4];
    const float* wv  = (const float*)d_in[5];
    const float* wo  = (const float*)d_in[6];
    float* out = (float*)d_out;

    const size_t NE_X = (size_t)MROWS * DMODEL;     // 8,388,608
    const size_t NE_W = (size_t)DMODEL * DMODEL;    // 4,194,304
    const size_t NE_Q = (size_t)BATCH * N_HEADS * SEQ * HEAD_DIM; // 8,388,608

    size_t off = 0;
    auto alloc = [&](size_t elems) { u16* p = (u16*)((char*)d_ws + off); off += elems * sizeof(u16); return p; };
    u16* xb  = alloc(NE_X);
    u16* wqb = alloc(NE_W);
    u16* wkb = alloc(NE_W);
    u16* wvb = alloc(NE_W);
    u16* wob = alloc(NE_W);
    u16* Qb  = alloc(NE_Q);
    u16* Kb  = alloc(NE_Q);
    u16* Vb  = alloc(NE_Q);
    u16* Ob  = alloc(NE_X);

    // 1) fp32 -> bf16 conversions
    cvt_bf16_kernel<<<8192, 256, 0, stream>>>(x,  xb,  (int)NE_X);
    cvt_bf16_kernel<<<4096, 256, 0, stream>>>(wq, wqb, (int)NE_W);
    cvt_bf16_kernel<<<4096, 256, 0, stream>>>(wk, wkb, (int)NE_W);
    cvt_bf16_kernel<<<4096, 256, 0, stream>>>(wv, wvb, (int)NE_W);
    cvt_bf16_kernel<<<4096, 256, 0, stream>>>(wo, wob, (int)NE_W);

    // 2) QKV projections (WMMA), store bf16 transposed to [B,H,S,HD]
    dim3 gq(DMODEL / 128, MROWS / 128, 3);
    gemm_bf16_kernel<<<gq, 256, 0, stream>>>(xb, wqb, wkb, wvb, Qb, Kb, Vb, nullptr, 0);

    // 3) RoPE in-place on Q,K
    rope_kernel<<<8192, 256, 0, stream>>>(Qb, Kb, fc);

    // 4) Flash attention with tanh softcap
    dim3 ga(SEQ / 128, BATCH * N_HEADS);
    attn_kernel<<<ga, 256, 0, stream>>>(Qb, Kb, Vb, Ob);

    // 5) Output projection -> fp32 d_out
    dim3 go(DMODEL / 128, MROWS / 128, 1);
    gemm_bf16_kernel<<<go, 256, 0, stream>>>(Ob, wob, wob, wob, nullptr, nullptr, nullptr, out, 1);
}
